// SWAttention_91122026152097
// MI455X (gfx1250) — compile-verified, matrix operand order checked by reading, not run on hardware
//
#include <hip/hip_runtime.h>
#include <hip/hip_bf16.h>
#include <stdint.h>

#define BB   2
#define TT   2048
#define HIDN 2048
#define NH   16
#define DD   128
#define QCC  512
#define WINW 512
#define GG   4
#define INTR 2048

typedef __attribute__((ext_vector_type(16))) __bf16 v16bf;
typedef __attribute__((ext_vector_type(8)))  float  v8f;

union BFrag { v16bf v; uint32_t u[8]; };

__device__ __forceinline__ uint16_t f2b(float f) {
  uint32_t x = __float_as_uint(f);
  x += 0x7FFFu + ((x >> 16) & 1u);          // round-to-nearest-even
  return (uint16_t)(x >> 16);
}
__device__ __forceinline__ uint32_t pk2(float a, float b) {
  return (uint32_t)f2b(a) | ((uint32_t)f2b(b) << 16);
}
__device__ __forceinline__ v8f v8f_zero() {
  v8f v;
#pragma unroll
  for (int i = 0; i < 8; ++i) v[i] = 0.0f;
  return v;
}

// Generic -> LDS(AS3) address: ptrtoint of an AS(3) pointer is the LDS byte
// offset, which is what the async-to-LDS VDST operand expects.
typedef __attribute__((address_space(3))) const void* lds_cvptr;
__device__ __forceinline__ uint32_t lds_off(const void* p) {
  return (uint32_t)(uintptr_t)(lds_cvptr)p;
}

// CDNA5 async copy: 16B global -> LDS, tracked by ASYNCcnt.
__device__ __forceinline__ void async_b128(uint32_t lds, uint64_t gaddr) {
  asm volatile("global_load_async_to_lds_b128 %0, %1, off"
               :: "v"(lds), "v"(gaddr) : "memory");
}
__device__ __forceinline__ void wait_async0() {
  asm volatile("s_wait_asynccnt 0" ::: "memory");
}

// Load a 16x32 bf16 A/B fragment (ISA 7.12.2 layout) from row-major storage.
__device__ __forceinline__ v16bf load_frag(const uint16_t* base, int stride) {
  const int lane = threadIdx.x & 31;
  const int m = lane & 15, hi = lane >> 4;
  const uint16_t* p = base + (size_t)m * stride + 8 * hi;
  uint4 a = *(const uint4*)(p);
  uint4 b = *(const uint4*)(p + 16);
  BFrag f;
  f.u[0] = a.x; f.u[1] = a.y; f.u[2] = a.z; f.u[3] = a.w;
  f.u[4] = b.x; f.u[5] = b.y; f.u[6] = b.z; f.u[7] = b.w;
  return f.v;
}

#define WMMA_BF16(A, B, C) \
  __builtin_amdgcn_wmma_f32_16x16x32_bf16(false, (A), false, (B), (short)0, (C), false, false)

// ---------------------------------------------------------------------------
// Tiled WMMA GEMM: C[M,N] = A[M,K] * B[K,N].  Block tile 128x128, BK=32,
// double-buffered LDS, 8 waves as 4(M)x2(N), each wave 32x64 (2x4 WMMA tiles).
// ATY: 0 = A f32 (convert to bf16 in regs), 1 = A bf16 (async copy to LDS)
// BTY: 0 = B f32 row-major (K,N) (convert+transpose in regs),
//      1 = B bf16 pre-transposed (N,K) (async copy to LDS)
// OTY: 0 = C f32, 1 = C bf16, 2 = C bf16 stored transposed (C^T[n*ldc + m])
// ---------------------------------------------------------------------------
template <int ATY, int BTY, int OTY>
__global__ __launch_bounds__(256) void gemm_wmma_k(
    const void* __restrict__ Ap, const void* __restrict__ Bp, void* __restrict__ Cp,
    int K, int lda, int ldb, int ldc,
    long long sA, long long sB, long long sC) {
  __shared__ __align__(16) uint16_t As[2][128 * 32];
  __shared__ __align__(16) uint16_t Bts[2][128 * 32];   // B^T tile: [n][k]

  const int tid = threadIdx.x;
  const int lane = tid & 31, wid = tid >> 5;
  const int wm = wid >> 1, wn = wid & 1;
  const long long z = blockIdx.z;
  const int gm0 = blockIdx.y * 128, gn0 = blockIdx.x * 128;

  const int arow = tid >> 1, acol = (tid & 1) << 4;   // A: 128 rows x 32 k
  const int bk = tid >> 3, bn = (tid & 7) << 4;       // B f32: 32 k x 128 n
  const int tn = tid >> 1, tkc = (tid & 1) << 4;      // B bf16T: 128 n x 32 k

  v8f acc[2][4];
#pragma unroll
  for (int i = 0; i < 2; ++i)
#pragma unroll
    for (int j = 0; j < 4; ++j) acc[i][j] = v8f_zero();

  float4 ra[4], rb[4];

  auto loadA_regs = [&](int kk) {
    const float* a = (const float*)Ap + (size_t)(z * sA) +
                     (size_t)(gm0 + arow) * lda + kk + acol;
    ra[0] = *(const float4*)(a);
    ra[1] = *(const float4*)(a + 4);
    ra[2] = *(const float4*)(a + 8);
    ra[3] = *(const float4*)(a + 12);
  };
  auto storeA_lds = [&](int nb) {
    uint32_t* dst = (uint32_t*)&As[nb][arow * 32 + acol];
    dst[0] = pk2(ra[0].x, ra[0].y); dst[1] = pk2(ra[0].z, ra[0].w);
    dst[2] = pk2(ra[1].x, ra[1].y); dst[3] = pk2(ra[1].z, ra[1].w);
    dst[4] = pk2(ra[2].x, ra[2].y); dst[5] = pk2(ra[2].z, ra[2].w);
    dst[6] = pk2(ra[3].x, ra[3].y); dst[7] = pk2(ra[3].z, ra[3].w);
  };
  auto asyncA = [&](int kk, int nb) {
    const uint16_t* a = (const uint16_t*)Ap + (size_t)(z * sA) +
                        (size_t)(gm0 + arow) * lda + kk + acol;
    uint32_t l0 = lds_off(&As[nb][arow * 32 + acol]);
    async_b128(l0,      (uint64_t)(uintptr_t)a);
    async_b128(l0 + 16, (uint64_t)(uintptr_t)(a + 8));
  };
  auto loadB_regs = [&](int kk) {
    const float* bs = (const float*)Bp + (size_t)(z * sB) +
                      (size_t)(kk + bk) * ldb + gn0 + bn;
    rb[0] = *(const float4*)(bs);
    rb[1] = *(const float4*)(bs + 4);
    rb[2] = *(const float4*)(bs + 8);
    rb[3] = *(const float4*)(bs + 12);
  };
  auto storeB_lds = [&](int nb) {
    uint16_t* d = &Bts[nb][0];
#pragma unroll
    for (int q = 0; q < 4; ++q) {
      d[(bn + q * 4 + 0) * 32 + bk] = f2b(rb[q].x);
      d[(bn + q * 4 + 1) * 32 + bk] = f2b(rb[q].y);
      d[(bn + q * 4 + 2) * 32 + bk] = f2b(rb[q].z);
      d[(bn + q * 4 + 3) * 32 + bk] = f2b(rb[q].w);
    }
  };
  auto asyncB = [&](int kk, int nb) {
    const uint16_t* bsrc = (const uint16_t*)Bp + (size_t)(z * sB) +
                           (size_t)(gn0 + tn) * ldb + kk + tkc;
    uint32_t l0 = lds_off(&Bts[nb][tn * 32 + tkc]);
    async_b128(l0,      (uint64_t)(uintptr_t)bsrc);
    async_b128(l0 + 16, (uint64_t)(uintptr_t)(bsrc + 8));
  };
  auto compute = [&](int cb) {
    const uint16_t* Ab = &As[cb][(wm * 32) * 32];
    const uint16_t* Bb = &Bts[cb][(wn * 64) * 32];
    v16bf a0 = load_frag(Ab, 32);
    v16bf a1 = load_frag(Ab + 512, 32);
#pragma unroll
    for (int wj = 0; wj < 4; ++wj) {
      v16bf bf = load_frag(Bb + wj * 512, 32);
      acc[0][wj] = WMMA_BF16(a0, bf, acc[0][wj]);
      acc[1][wj] = WMMA_BF16(a1, bf, acc[1][wj]);
    }
  };

  // prologue: stage tile 0
  if (ATY == 0) { loadA_regs(0); } else { asyncA(0, 0); }
  if (BTY == 0) { loadB_regs(0); } else { asyncB(0, 0); }
  if (ATY == 0) storeA_lds(0);
  if (BTY == 0) storeB_lds(0);
  if (ATY == 1 || BTY == 1) wait_async0();
  __syncthreads();

  int buf = 0;
  for (int k0 = 0; k0 < K; k0 += 32) {
    const bool nxt = (k0 + 32) < K;
    if (nxt) {
      if (ATY == 0) loadA_regs(k0 + 32); else asyncA(k0 + 32, buf ^ 1);
      if (BTY == 0) loadB_regs(k0 + 32); else asyncB(k0 + 32, buf ^ 1);
    }
    compute(buf);
    if (nxt) {
      if (ATY == 0) storeA_lds(buf ^ 1);
      if (BTY == 0) storeB_lds(buf ^ 1);
      if (ATY == 1 || BTY == 1) wait_async0();
    }
    __syncthreads();
    buf ^= 1;
  }

  // writeback (C frag: VGPR j -> row j+8*hi, lane&15 -> col)
  const int n = lane & 15, hi = lane >> 4;
  auto wb = [&](v8f a, int wi, int wj) {
#pragma unroll
    for (int j = 0; j < 8; ++j) {
      float val = a[j];
      size_t row = (size_t)(gm0 + wm * 32 + wi * 16 + 8 * hi + j);
      size_t col = (size_t)(gn0 + wn * 64 + wj * 16 + n);
      if (OTY == 0) {
        ((float*)Cp)[(size_t)(z * sC) + row * ldc + col] = val;
      } else if (OTY == 1) {
        ((uint16_t*)Cp)[(size_t)(z * sC) + row * ldc + col] = f2b(val);
      } else {
        ((uint16_t*)Cp)[(size_t)(z * sC) + col * ldc + row] = f2b(val);
      }
    }
  };
#pragma unroll
  for (int wi = 0; wi < 2; ++wi)
#pragma unroll
    for (int wj = 0; wj < 4; ++wj) wb(acc[wi][wj], wi, wj);
}

// ---------------------------------------------------------------------------
// One-time transpose+convert: in (K,N) f32 -> out (N,K) bf16.
// ---------------------------------------------------------------------------
__global__ __launch_bounds__(256) void transpose_bf16_k(
    const float* __restrict__ in, uint16_t* __restrict__ out, int K, int N) {
  __shared__ float tile[32][33];
  const int kx = blockIdx.x * 32, ny = blockIdx.y * 32;
  const int tx = threadIdx.x & 31, ty = threadIdx.x >> 5;
#pragma unroll
  for (int i = ty; i < 32; i += 8)
    tile[i][tx] = in[(size_t)(kx + i) * N + ny + tx];
  __syncthreads();
#pragma unroll
  for (int i = ty; i < 32; i += 8)
    out[(size_t)(ny + i) * K + kx + tx] = f2b(tile[tx][i]);
}

// ---------------------------------------------------------------------------
// Partial RoPE (first 64 dims, pairs (i, i+32)) + RMSNorm over 128, f32->bf16.
// ---------------------------------------------------------------------------
__global__ __launch_bounds__(256) void rope_norm_kernel(
    const float* __restrict__ in, uint16_t* __restrict__ out,
    const float* __restrict__ w, int nvec, int vecs_per_t) {
  const int wid = threadIdx.x >> 5, lane = threadIdx.x & 31;
  const int v = blockIdx.x * 8 + wid;
  if (v >= nvec) return;
  const int t = (v / vecs_per_t) % TT;
  const float* x = in + (size_t)v * 128;
  float x1 = x[lane], x2 = x[lane + 32], p1 = x[lane + 64], p2 = x[lane + 96];
  float ang = (float)t * __expf(-0.28782313662425574f * (float)lane);
  float s, c;
  __sincosf(ang, &s, &c);
  float r1 = x1 * c - x2 * s;
  float r2 = x1 * s + x2 * c;
  float ss = r1 * r1 + r2 * r2 + p1 * p1 + p2 * p2;
#pragma unroll
  for (int m = 1; m < 32; m <<= 1) ss += __shfl_xor(ss, m, 32);
  float rms = rsqrtf(ss * (1.0f / 128.0f) + 1e-6f);
  uint16_t* o = out + (size_t)v * 128;
  o[lane]      = f2b(r1 * rms * w[lane]);
  o[lane + 32] = f2b(r2 * rms * w[lane + 32]);
  o[lane + 64] = f2b(p1 * rms * w[lane + 64]);
  o[lane + 96] = f2b(p2 * rms * w[lane + 96]);
}

// ---------------------------------------------------------------------------
// Sliding-window flash attention with sink logit (one wave per (b,h,16 rows)).
// ---------------------------------------------------------------------------
__global__ __launch_bounds__(256) void attn_kernel(
    const uint16_t* __restrict__ qb, const uint16_t* __restrict__ kb,
    const uint16_t* __restrict__ vt, const float* __restrict__ sink,
    float* __restrict__ xout) {
  __shared__ __align__(16) uint16_t pbuf[8][16 * 32];
  const int lane = threadIdx.x & 31;
  const int wid  = threadIdx.x >> 5;
  const int gid  = blockIdx.x * 8 + wid;
  const int qt = gid & 127;
  const int h  = (gid >> 7) & 15;
  const int b  = gid >> 11;
  const int t0 = qt * 16;
  const int m  = lane & 15;
  const int hi = lane >> 4;

  v16bf aq[4];
  {
    const uint16_t* qrow = qb + (size_t)(b * TT + t0) * (NH * DD) + h * DD;
#pragma unroll
    for (int kc = 0; kc < 4; ++kc) aq[kc] = load_frag(qrow + kc * 32, NH * DD);
  }

  float mrun[8], lrun[8];
  const float slog = sink[h];
#pragma unroll
  for (int j = 0; j < 8; ++j) { mrun[j] = slog; lrun[j] = 1.0f; }
  v8f acc[8];
#pragma unroll
  for (int dt = 0; dt < 8; ++dt) acc[dt] = v8f_zero();

  const float scale = 0.088388347648318447f;   // 1/sqrt(128)
  int s_lo = t0 - (WINW - 1);
  if (s_lo < 0) s_lo = 0;
  s_lo &= ~31;

  uint16_t* pb = &pbuf[wid][0];

  for (int sc = s_lo; sc <= t0 + 15; sc += 32) {
    if (sc + 32 <= t0 + 15)    // prefetch next K chunk (global_prefetch_b8)
      __builtin_prefetch(kb + (size_t)(b * TT + sc + 32) * DD + lane * 8, 0, 1);
    // scores: Q(16xD) @ K^T(Dx32) as two 16x16 tiles
    v8f s0 = v8f_zero(), s1 = v8f_zero();
    const uint16_t* krow0 = kb + (size_t)(b * TT + sc) * DD;
    const uint16_t* krow1 = kb + (size_t)(b * TT + sc + 16) * DD;
#pragma unroll
    for (int kc = 0; kc < 4; ++kc) {
      v16bf bk0 = load_frag(krow0 + kc * 32, DD);
      v16bf bk1 = load_frag(krow1 + kc * 32, DD);
      s0 = WMMA_BF16(aq[kc], bk0, s0);
      s1 = WMMA_BF16(aq[kc], bk1, s1);
    }
    // mask + online softmax (row j+8*hi spread over 16 lanes)
#pragma unroll
    for (int j = 0; j < 8; ++j) {
      int qpos = t0 + j + 8 * hi;
      int k0p = sc + m, k1p = sc + 16 + m;
      float v0 = ((unsigned)(qpos - k0p) < WINW) ? s0[j] * scale : -1e30f;
      float v1 = ((unsigned)(qpos - k1p) < WINW) ? s1[j] * scale : -1e30f;
      float mx = fmaxf(v0, v1);
#pragma unroll
      for (int msk = 1; msk < 16; msk <<= 1) mx = fmaxf(mx, __shfl_xor(mx, msk, 16));
      float mn = fmaxf(mrun[j], mx);
      float corr = __expf(mrun[j] - mn);
      float pr0 = __expf(v0 - mn);
      float pr1 = __expf(v1 - mn);
      float rs = pr0 + pr1;
#pragma unroll
      for (int msk = 1; msk < 16; msk <<= 1) rs += __shfl_xor(rs, msk, 16);
      lrun[j] = lrun[j] * corr + rs;
      mrun[j] = mn;
#pragma unroll
      for (int dt = 0; dt < 8; ++dt) acc[dt][j] *= corr;
      pb[(j + 8 * hi) * 32 + m]      = f2b(pr0);
      pb[(j + 8 * hi) * 32 + 16 + m] = f2b(pr1);
    }
    asm volatile("s_wait_dscnt 0" ::: "memory");
    // PV: P(16x32) @ V(32x128); V fragments from transposed V (contiguous in k)
    v16bf ap = load_frag(pb, 32);
    const uint16_t* vbase = vt + (size_t)(b * TT + sc);
#pragma unroll
    for (int dt = 0; dt < 8; ++dt) {
      v16bf bv = load_frag(vbase + (size_t)(dt * 16) * (BB * TT), BB * TT);
      acc[dt] = WMMA_BF16(ap, bv, acc[dt]);
    }
  }

#pragma unroll
  for (int dt = 0; dt < 8; ++dt) {
#pragma unroll
    for (int j = 0; j < 8; ++j) {
      size_t row = (size_t)(b * TT + t0 + j + 8 * hi);
      xout[row * (NH * DD) + h * DD + dt * 16 + m] = acc[dt][j] / lrun[j];
    }
  }
}

// ---------------------------------------------------------------------------
extern "C" void kernel_launch(void* const* d_in, const int* in_sizes, int n_in,
                              void* d_out, int out_size, void* d_ws, size_t ws_size,
                              hipStream_t stream) {
  (void)in_sizes; (void)n_in; (void)out_size; (void)ws_size;
  const float* h     = (const float*)d_in[0];
  const float* wq_c  = (const float*)d_in[1];
  const float* wq_u  = (const float*)d_in[2];
  const float* wk    = (const float*)d_in[3];
  const float* wv    = (const float*)d_in[4];
  const float* qnw   = (const float*)d_in[5];
  const float* knw   = (const float*)d_in[6];
  const float* sinkl = (const float*)d_in[7];
  const float* w1    = (const float*)d_in[8];
  const float* w2    = (const float*)d_in[9];
  float* out = (float*)d_out;

  char* ws = (char*)d_ws;
  const int MT = BB * TT;  // 4096
  uint16_t* qc16 = (uint16_t*)(ws);                 //  4 MB  q_comp bf16 (4096x512)
  float*    qf32 = (float*)(ws + 4194304);          // 32 MB  q f32 / later attn out x
  uint16_t* qb16 = (uint16_t*)(ws + 37748736);      // 16 MB  q bf16 (4096x2048)
  float*    kraw = (float*)(ws + 54525952);         //  2 MB  k pre-norm f32 (4096x128)
  uint16_t* kb16 = (uint16_t*)(ws + 56623104);      //  1 MB  k bf16
  uint16_t* vt16 = (uint16_t*)(ws + 57671680);      //  1 MB  v^T bf16 (128x4096)
  uint16_t* y16  = (uint16_t*)(ws + 58720256);      // 64 MB  MLP mid bf16 (4096x8192)
  uint16_t* w2t  = (uint16_t*)(ws + 125829120);     // 32 MB  out_w2^T bf16 (2048x8192)

  // 0. one-time: out_w2 (8192,2048) f32 -> (2048,8192) bf16 transposed
  transpose_bf16_k<<<dim3((GG * INTR) / 32, HIDN / 32, 1), 256, 0, stream>>>(
      w2, w2t, GG * INTR, HIDN);
  // 1. q_comp = h @ wq_comp  -> bf16
  gemm_wmma_k<0, 0, 1><<<dim3(QCC / 128, MT / 128, 1), 256, 0, stream>>>(
      h, wq_c, qc16, HIDN, HIDN, QCC, QCC, 0, 0, 0);
  // 2. k_raw = h @ wk  -> f32
  gemm_wmma_k<0, 0, 0><<<dim3(DD / 128, MT / 128, 1), 256, 0, stream>>>(
      h, wk, kraw, HIDN, HIDN, DD, DD, 0, 0, 0);
  // 3. v^T = (h @ wv)^T  -> bf16 transposed (ldc = MT)
  gemm_wmma_k<0, 0, 2><<<dim3(DD / 128, MT / 128, 1), 256, 0, stream>>>(
      h, wv, vt16, HIDN, HIDN, DD, MT, 0, 0, 0);
  // 4. q = q_comp @ wq_up  -> f32  (A bf16: async-to-LDS staging)
  gemm_wmma_k<1, 0, 0><<<dim3((NH * DD) / 128, MT / 128, 1), 256, 0, stream>>>(
      qc16, wq_u, qf32, QCC, QCC, NH * DD, NH * DD, 0, 0, 0);
  // 5. rope + rmsnorm on q -> bf16
  rope_norm_kernel<<<dim3(MT * NH / 8), 256, 0, stream>>>(qf32, qb16, qnw, MT * NH, NH);
  // 6. rope + rmsnorm on k -> bf16
  rope_norm_kernel<<<dim3(MT / 8), 256, 0, stream>>>(kraw, kb16, knw, MT, 1);
  // 7. sliding-window attention with sink -> x (reuses qf32 buffer)
  attn_kernel<<<dim3(BB * NH * (TT / 16) / 8), 256, 0, stream>>>(
      qb16, kb16, vt16, sinkl, qf32);
  // 8. grouped MLP1: y[:, g*2048:] = x[:, g*512:] @ w1[g] -> bf16
  gemm_wmma_k<0, 0, 1><<<dim3(INTR / 128, MT / 128, GG), 256, 0, stream>>>(
      qf32, w1, y16, QCC, NH * DD, INTR, GG * INTR,
      (long long)512, (long long)512 * 2048, (long long)2048);
  // 9. out = y @ out_w2 -> f32 (A and B both bf16: fully async-staged)
  gemm_wmma_k<1, 1, 0><<<dim3(HIDN / 128, MT / 128, 1), 256, 0, stream>>>(
      y16, w2t, out, GG * INTR, GG * INTR, GG * INTR, HIDN, 0, 0, 0);
}
